// Generator_54743653154966
// MI455X (gfx1250) — compile-verified, matrix operand order checked by reading, not run on hardware
//
#include <hip/hip_runtime.h>

#define N_NODES 8192
#define F_DIM   512
#define H_DIM   128
#define E_EDGES 262144

typedef float    v2f  __attribute__((ext_vector_type(2)));
typedef float    v4f  __attribute__((ext_vector_type(4)));
typedef float    v8f  __attribute__((ext_vector_type(8)));
typedef _Float16 v16h __attribute__((ext_vector_type(16)));

// ---------- ordered-uint encoding for float atomic min/max ----------
__device__ __forceinline__ unsigned enc_f(float f) {
  unsigned u = __float_as_uint(f);
  return (u & 0x80000000u) ? ~u : (u | 0x80000000u);
}
__device__ __forceinline__ float dec_f(unsigned e) {
  unsigned u = (e & 0x80000000u) ? (e & 0x7FFFFFFFu) : ~e;
  return __uint_as_float(u);
}

// ---------- 1) feature = where(mask_u < 0.2, 0, x) ----------
__global__ __launch_bounds__(256) void feature_k(const v4f* __restrict__ x4,
                                                 const float* __restrict__ mask_u,
                                                 v4f* __restrict__ out4) {
  int i  = blockIdx.x * 256 + threadIdx.x;       // N*F/4 threads
  int c4 = i & (F_DIM / 4 - 1);
  int c  = c4 * 4;
  v4f v = x4[i];
  v.x = (mask_u[c + 0] < 0.2f) ? 0.0f : v.x;
  v.y = (mask_u[c + 1] < 0.2f) ? 0.0f : v.y;
  v.z = (mask_u[c + 2] < 0.2f) ? 0.0f : v.z;
  v.w = (mask_u[c + 3] < 0.2f) ? 0.0f : v.w;
  out4[i] = v;
}

// ---------- 2) streaming zero fill (non-temporal B128 stores) ----------
__global__ __launch_bounds__(256) void zero4_k(v4f* __restrict__ p, size_t n4) {
  size_t i      = (size_t)blockIdx.x * blockDim.x + threadIdx.x;
  size_t stride = (size_t)gridDim.x * blockDim.x;
  v4f z = {0.0f, 0.0f, 0.0f, 0.0f};
  for (; i < n4; i += stride) __builtin_nontemporal_store(z, &p[i]);
}

__global__ void initctl_k(unsigned* ctl) {
  ctl[0] = 0xFFFFFFFFu;  // min (encoded)
  ctl[1] = 0u;           // max (encoded)
  ctl[2] = 0u;           // duplicate edge count
  ctl[3] = 0u;           // nnz(adj)
}

// ---------- 3) xw = x @ W_enc via CDNA5 WMMA ----------
// One wave owns a 16-row M-tile and all 128 output columns (8 accumulators).
__global__ __launch_bounds__(256) void gemm_k(const float* __restrict__ x,
                                              const float* __restrict__ W,
                                              float* __restrict__ xw) {
  int wave  = threadIdx.x >> 5;
  int lane  = threadIdx.x & 31;
  int half  = lane >> 4;       // 0: lanes 0-15, 1: lanes 16-31
  int l16   = lane & 15;
  int tileM = blockIdx.x * 8 + wave;   // 512 M-tiles total
  int m0    = tileM * 16;

  v8f acc[8] = {};
  const float* xrow = x + (size_t)(m0 + l16) * F_DIM;

#if __has_builtin(__builtin_amdgcn_wmma_f32_16x16x4_f32)
  // Native fp32 WMMA: A 16x4 (lane<16: K=k,k+1 ; lane>=16: K=k+2,k+3)
  for (int k = 0; k < F_DIM; k += 4) {
    v2f a;
    a.x = xrow[k + 2 * half + 0];
    a.y = xrow[k + 2 * half + 1];
#pragma unroll
    for (int j = 0; j < 8; ++j) {
      int n = j * 16 + l16;
      v2f b;
      b.x = W[(size_t)(k + 2 * half + 0) * H_DIM + n];
      b.y = W[(size_t)(k + 2 * half + 1) * H_DIM + n];
      acc[j] = __builtin_amdgcn_wmma_f32_16x16x4_f32(
          false, a, false, b, (short)0, acc[j], false, false);
    }
  }
#else
  // Fallback: convert to f16 and use 16x16x32 (layout per ISA 7.12.2)
  for (int k = 0; k < F_DIM; k += 32) {
    v16h a;
#pragma unroll
    for (int i = 0; i < 8; ++i) {
      a[i]     = (_Float16)xrow[k + 8 * half + i];
      a[8 + i] = (_Float16)xrow[k + 16 + 8 * half + i];
    }
#pragma unroll
    for (int j = 0; j < 8; ++j) {
      int n = j * 16 + l16;
      v16h b;
#pragma unroll
      for (int i = 0; i < 8; ++i) {
        b[i]     = (_Float16)W[(size_t)(k + 8 * half + i) * H_DIM + n];
        b[8 + i] = (_Float16)W[(size_t)(k + 16 + 8 * half + i) * H_DIM + n];
      }
      acc[j] = __builtin_amdgcn_wmma_f32_16x16x32_f16(
          false, a, false, b, (short)0, acc[j], false, false);
    }
  }
#endif

  // C/D layout: VGPR r -> rows m0+r (lanes 0-15) / m0+r+8 (lanes 16-31)
#pragma unroll
  for (int j = 0; j < 8; ++j) {
    int n = j * 16 + l16;
#pragma unroll
    for (int r = 0; r < 8; ++r) {
      int m = m0 + r + half * 8;
      xw[(size_t)m * H_DIM + n] = acc[j][r];
    }
  }
}

// ---------- 4) agg[row] += xw[col]  (segment_sum) ----------
__global__ __launch_bounds__(256) void scatter_add_k(const float* __restrict__ xw,
                                                     const int* __restrict__ row,
                                                     const int* __restrict__ col,
                                                     float* __restrict__ agg) {
  int e = blockIdx.x * 2 + (threadIdx.x >> 7);   // 2 edges / block
  int h = threadIdx.x & (H_DIM - 1);
  int r = row[e], c = col[e];
  atomicAdd(&agg[(size_t)r * H_DIM + h], xw[(size_t)c * H_DIM + h]);
}

// ---------- 5) s1[n] = relu(agg[n])·W1 ; s2[n] = relu(agg[n])·W2 ----------
__global__ __launch_bounds__(256) void nodes_k(const float* __restrict__ agg,
                                               const float* __restrict__ W_edge,
                                               float* __restrict__ s1,
                                               float* __restrict__ s2) {
  int n = blockIdx.x * 256 + threadIdx.x;
  if (n >= N_NODES) return;
  const float* e = agg + (size_t)n * H_DIM;
  float a = 0.0f, b = 0.0f;
#pragma unroll 4
  for (int h = 0; h < H_DIM; ++h) {
    float v = fmaxf(e[h], 0.0f);
    a += v * W_edge[h];
    b += v * W_edge[H_DIM + h];
  }
  s1[n] = a;
  s2[n] = b;
}

// ---------- 6) edge logits + global min/max ----------
__global__ __launch_bounds__(256) void elog_k(const float* __restrict__ s1,
                                              const float* __restrict__ s2,
                                              const int* __restrict__ row,
                                              const int* __restrict__ col,
                                              const float* __restrict__ b_edge,
                                              float* __restrict__ logits,
                                              unsigned* __restrict__ ctl) {
  int e = blockIdx.x * 256 + threadIdx.x;        // E divisible by 256
  float l = s1[row[e]] + s2[col[e]] + b_edge[0];
  logits[e] = l;
  float mn = l, mx = l;
#pragma unroll
  for (int o = 16; o > 0; o >>= 1) {
    mn = fminf(mn, __shfl_xor(mn, o, 32));
    mx = fmaxf(mx, __shfl_xor(mx, o, 32));
  }
  if ((threadIdx.x & 31) == 0) {
    atomicMin(&ctl[0], enc_f(mn));
    atomicMax(&ctl[1], enc_f(mx));
  }
}

// ---------- 7) mark edge positions; count duplicates exactly ----------
__global__ __launch_bounds__(256) void mark_k(const int* __restrict__ row,
                                              const int* __restrict__ col,
                                              float* __restrict__ adj,
                                              unsigned* __restrict__ ctl) {
  int e = blockIdx.x * 256 + threadIdx.x;
  size_t pos = (size_t)row[e] * N_NODES + col[e];
  float old = atomicExch(&adj[pos], 1.0f);
  if (old != 0.0f) atomicAdd(&ctl[2], 1u);
}

// ---------- 8) final per-edge Gumbel/concrete value into adj ----------
__global__ __launch_bounds__(256) void fin_edge_k(const int* __restrict__ row,
                                                  const int* __restrict__ col,
                                                  const float* __restrict__ logits,
                                                  const float* __restrict__ noise,
                                                  const unsigned* __restrict__ ctl,
                                                  float* __restrict__ adj) {
  int e = blockIdx.x * 256 + threadIdx.x;
  float lo = dec_f(ctl[0]);
  float hi = dec_f(ctl[1]);
  float k  = 0.3f / (hi - lo);                   // ub=0.3, lb=0
  float p  = k * (logits[e] - lo);
  size_t pos = (size_t)row[e] * N_NODES + col[e];
  float u  = noise[pos];
  float lp = logf(p + 1e-10f) - log1pf(-p + 1e-10f);
  float lg = logf(u) - log1pf(-u);
  float z  = (lp + lg) * 5.0f;                   // / TEMPERATURE (0.2)
  float soft = 1.0f / (1.0f + expf(-z));
  float hard = rintf(soft);                      // round-half-even == jnp.round
  adj[pos] = 1.0f - hard;                        // hard + soft - sg(soft) == hard
}

// ---------- 9) nnz(adj) via streaming scan ----------
__global__ __launch_bounds__(256) void count_k(const v4f* __restrict__ p, size_t n4,
                                               unsigned* __restrict__ ctl) {
  size_t i      = (size_t)blockIdx.x * blockDim.x + threadIdx.x;
  size_t stride = (size_t)gridDim.x * blockDim.x;
  unsigned c = 0;
  for (; i < n4; i += stride) {
    v4f v = __builtin_nontemporal_load(&p[i]);
    c += (v.x != 0.0f) + (v.y != 0.0f) + (v.z != 0.0f) + (v.w != 0.0f);
  }
#pragma unroll
  for (int o = 16; o > 0; o >>= 1) c += __shfl_xor(c, o, 32);
  if ((threadIdx.x & 31) == 0) atomicAdd(&ctl[3], c);
}

// ---------- 10) rate = nnz(adj) / distinct(edges) ----------
__global__ void rate_k(const unsigned* __restrict__ ctl, float* __restrict__ out) {
  out[0] = (float)ctl[3] / (float)(E_EDGES - ctl[2]);
}

extern "C" void kernel_launch(void* const* d_in, const int* in_sizes, int n_in,
                              void* d_out, int out_size, void* d_ws, size_t ws_size,
                              hipStream_t stream) {
  const float* x      = (const float*)d_in[0];
  const int*   row    = (const int*)d_in[1];
  const int*   col    = (const int*)d_in[2];
  const float* W_enc  = (const float*)d_in[3];
  const float* W_edge = (const float*)d_in[4];
  const float* b_edge = (const float*)d_in[5];
  const float* noise  = (const float*)d_in[6];
  const float* mask_u = (const float*)d_in[7];

  float* out     = (float*)d_out;
  float* feature = out;
  float* adj     = out + (size_t)N_NODES * F_DIM;
  float* rate    = adj + (size_t)N_NODES * N_NODES;

  float*    ws     = (float*)d_ws;
  float*    xw     = ws;
  float*    agg    = xw + (size_t)N_NODES * H_DIM;
  float*    s1     = agg + (size_t)N_NODES * H_DIM;
  float*    s2     = s1 + N_NODES;
  float*    logits = s2 + N_NODES;
  unsigned* ctl    = (unsigned*)(logits + E_EDGES);

  const size_t NN  = (size_t)N_NODES * N_NODES;

  // feature mask (N*F/4 vec4 threads)
  feature_k<<<(N_NODES * F_DIM / 4) / 256, 256, 0, stream>>>((const v4f*)x, mask_u, (v4f*)feature);
  // zero adj (268 MB, NT stores) and agg (4 MB)
  zero4_k<<<8192, 256, 0, stream>>>((v4f*)adj, NN / 4);
  zero4_k<<<256, 256, 0, stream>>>((v4f*)agg, (size_t)N_NODES * H_DIM / 4);
  initctl_k<<<1, 1, 0, stream>>>(ctl);
  // GEMM: 512 M-tiles, 8 waves/block
  gemm_k<<<64, 256, 0, stream>>>(x, W_enc, xw);
  // segment_sum
  scatter_add_k<<<E_EDGES / 2, 256, 0, stream>>>(xw, row, col, agg);
  // per-node dot products
  nodes_k<<<N_NODES / 256, 256, 0, stream>>>(agg, W_edge, s1, s2);
  // edge logits + min/max
  elog_k<<<E_EDGES / 256, 256, 0, stream>>>(s1, s2, row, col, b_edge, logits, ctl);
  // mark positions + exact duplicate count
  mark_k<<<E_EDGES / 256, 256, 0, stream>>>(row, col, adj, ctl);
  // final edge values
  fin_edge_k<<<E_EDGES / 256, 256, 0, stream>>>(row, col, logits, noise, ctl, adj);
  // nnz scan
  count_k<<<8192, 256, 0, stream>>>((const v4f*)adj, NN / 4, ctl);
  // scalar output
  rate_k<<<1, 1, 0, stream>>>(ctl, rate);
}
